// MultiHeadAttentionQuantum_65481071402573
// MI455X (gfx1250) — compile-verified, matrix operand order checked by reading, not run on hardware
//
#include <hip/hip_runtime.h>
#include <stdint.h>

typedef __attribute__((ext_vector_type(16))) _Float16 v16h;
typedef __attribute__((ext_vector_type(8)))  _Float16 v8h;
typedef __attribute__((ext_vector_type(4)))  _Float16 v4h;
typedef __attribute__((ext_vector_type(8)))  float    v8f;

#define TM 128
#define TN 128
#define TK 32
#define LDSW 40   // padded LDS row stride in halves (80 B, keeps 16 B alignment)

// ---- one-shot W (f32) -> Wh (f16) conversion into workspace ----
__global__ __launch_bounds__(256)
void w_to_f16_kernel(const float* __restrict__ W, _Float16* __restrict__ Wh, int n4) {
    int i = blockIdx.x * blockDim.x + threadIdx.x;
    if (i < n4) {
        float4 v = ((const float4*)W)[i];
        v4h h;
        h[0] = (_Float16)v.x; h[1] = (_Float16)v.y;
        h[2] = (_Float16)v.z; h[3] = (_Float16)v.w;
        ((v4h*)Wh)[i] = h;
    }
}

// out[m,n] = sum_k cos(x[m,k] + theta[k&31]) * W[n,k] + bias[n]
// M = 32768, N = K = 1024. 128x128 tile / block, 8 waves, 4x2 WMMA subtiles / wave.
template<bool ASYNC_B>
__global__ __launch_bounds__(256)
void qmha_wmma_kernel(const float* __restrict__ x,
                      const float* __restrict__ theta,
                      const float* __restrict__ Wf32,
                      const _Float16* __restrict__ Wh,
                      const float* __restrict__ bias,
                      float* __restrict__ out,
                      int M, int N, int K)
{
    __shared__ __align__(16) _Float16 As[2][TM * LDSW];
    __shared__ __align__(16) _Float16 Bs[2][TN * LDSW];
    __shared__ float th[32];

    const int tid   = threadIdx.x;
    const int lane  = tid & 31;
    const int wave  = tid >> 5;     // 0..7
    const int waveM = wave & 1;     // 2 waves along M -> 64 rows each
    const int waveN = wave >> 1;    // 4 waves along N -> 32 cols each

    // N tiles on fastest grid dim: concurrent column blocks at the same m0
    // share x cachelines through L2 (192 MB) -> x streamed from HBM ~once.
    const int n0 = blockIdx.x * TN;
    const int m0 = blockIdx.y * TM;

    if (tid < 32) th[tid] = theta[tid];
    __syncthreads();

    v8f acc[4][2] = {};

    const int lr  = tid >> 1;          // staging row 0..127
    const int lc4 = (tid & 1) * 4;     // float4 chunk base (0 or 4)
    const size_t xrow = (size_t)(m0 + lr) * K;
    const size_t wrow = (size_t)(n0 + lr) * K;

    // ---- stage A = cos(x + theta) -> f16 into LDS buffer `buf` ----
    auto stageA = [&](int buf, int k0) {
        const float4* xr = (const float4*)(x + xrow + k0);
        #pragma unroll
        for (int i = 0; i < 4; ++i) {
            float4 v = xr[lc4 + i];
            int kk = (lc4 + i) * 4;                 // theta period 32 == TK
            v4h h;
            h[0] = (_Float16)__cosf(v.x + th[kk + 0]);
            h[1] = (_Float16)__cosf(v.y + th[kk + 1]);
            h[2] = (_Float16)__cosf(v.z + th[kk + 2]);
            h[3] = (_Float16)__cosf(v.w + th[kk + 3]);
            *(v4h*)&As[buf][lr * LDSW + kk] = h;
        }
    };

    // ---- stage B tile ----
    auto stageB = [&](int buf, int k0) {
        if constexpr (ASYNC_B) {
            // pure f16 byte-copy: async global->LDS DMA, 16 B per lane,
            // 512 chunks (128 rows x 4 chunks), 2 per thread. ASYNCcnt-tracked.
            #pragma unroll
            for (int t = 0; t < 2; ++t) {
                int idx = tid + t * 256;           // 0..511
                int row = idx >> 2;                // 0..127
                int ch  = idx & 3;                 // 16B chunk in row
                const _Float16* g = Wh + (size_t)(n0 + row) * K + k0 + ch * 8;
                uint32_t lds = (uint32_t)(uintptr_t)&Bs[buf][row * LDSW + ch * 8];
                asm volatile("global_load_async_to_lds_b128 %0, %1, off"
                             :: "v"(lds), "v"(g) : "memory");
            }
        } else {
            const float4* wr = (const float4*)(Wf32 + wrow + k0);
            #pragma unroll
            for (int i = 0; i < 4; ++i) {
                float4 v = wr[lc4 + i];
                int kk = (lc4 + i) * 4;
                v4h h;
                h[0] = (_Float16)v.x; h[1] = (_Float16)v.y;
                h[2] = (_Float16)v.z; h[3] = (_Float16)v.w;
                *(v4h*)&Bs[buf][lr * LDSW + kk] = h;
            }
        }
    };

    // ---- WMMA compute on buffer `buf` ----
    // ISA 16-bit A layout: lanes 0-15 carry K {0..7,16..23} of row M=lane&15,
    // lanes 16-31 carry K {8..15,24..31}; B mirrored (column = lane&15).
    auto compute = [&](int buf) {
        const int koff = (lane < 16) ? 0 : 8;
        const int l16  = lane & 15;
        v16h bfrag[2];
        #pragma unroll
        for (int ni = 0; ni < 2; ++ni) {
            const _Float16* src = &Bs[buf][(waveN * 32 + ni * 16 + l16) * LDSW];
            v8h lo = *(const v8h*)(src + koff);
            v8h hi = *(const v8h*)(src + koff + 16);
            #pragma unroll
            for (int j = 0; j < 8; ++j) { bfrag[ni][j] = lo[j]; bfrag[ni][8 + j] = hi[j]; }
        }
        #pragma unroll
        for (int mi = 0; mi < 4; ++mi) {
            const _Float16* src = &As[buf][(waveM * 64 + mi * 16 + l16) * LDSW];
            v8h lo = *(const v8h*)(src + koff);
            v8h hi = *(const v8h*)(src + koff + 16);
            v16h afrag;
            #pragma unroll
            for (int j = 0; j < 8; ++j) { afrag[j] = lo[j]; afrag[8 + j] = hi[j]; }
            #pragma unroll
            for (int ni = 0; ni < 2; ++ni) {
                acc[mi][ni] = __builtin_amdgcn_wmma_f32_16x16x32_f16(
                    false, afrag, false, bfrag[ni],
                    (short)0, acc[mi][ni], false, false);
            }
        }
    };

    // ---- double-buffered pipeline: stage k+1 while computing k ----
    stageA(0, 0);
    stageB(0, 0);
    const int KT = K / TK;
    for (int kt = 0; kt < KT; ++kt) {
        if constexpr (ASYNC_B)
            asm volatile("s_wait_asynccnt 0" ::: "memory");   // B DMA landed in LDS
        __syncthreads();                                      // publish buf kt&1
        if (kt + 1 < KT) {
            int nb = (kt + 1) & 1;
            int k0 = (kt + 1) * TK;
            __builtin_prefetch(x + xrow + k0, 0, 1);          // global_prefetch_b8
            stageA(nb, k0);
            stageB(nb, k0);
        }
        compute(kt & 1);
    }

    // ---- epilogue: C/D layout -> VGPR r holds M=r (lanes 0-15) / M=8+r
    // (lanes 16-31), lane&15 selects N. Add bias, store f32. ----
    const int l16  = lane & 15;
    const int rofs = (lane >> 4) * 8;
    #pragma unroll
    for (int ni = 0; ni < 2; ++ni) {
        int n = n0 + waveN * 32 + ni * 16 + l16;
        float bv = bias[n];
        #pragma unroll
        for (int mi = 0; mi < 4; ++mi) {
            int mbase = m0 + waveM * 64 + mi * 16 + rofs;
            #pragma unroll
            for (int r = 0; r < 8; ++r) {
                out[(size_t)(mbase + r) * N + n] = acc[mi][ni][r] + bv;
            }
        }
    }
}

extern "C" void kernel_launch(void* const* d_in, const int* in_sizes, int n_in,
                              void* d_out, int out_size, void* d_ws, size_t ws_size,
                              hipStream_t stream) {
    (void)n_in; (void)out_size;
    const float* x     = (const float*)d_in[0];   // (B,T,E) f32
    const float* theta = (const float*)d_in[1];   // (32,)   f32
    const float* W     = (const float*)d_in[2];   // (E,E)   f32
    const float* bias  = (const float*)d_in[3];   // (E,)    f32
    float* out = (float*)d_out;

    const int E = in_sizes[3];           // 1024
    const int M = in_sizes[0] / E;       // B*T = 32768
    const int N = E;
    const int K = E;

    dim3 grid(N / TN, M / TM);           // (8, 256): N fastest for L2 reuse of x
    dim3 block(256);

    const size_t whBytes = (size_t)E * K * sizeof(_Float16);   // 2 MB
    if (ws_size >= whBytes) {
        _Float16* Wh = (_Float16*)d_ws;
        int n4 = (E * K) / 4;
        w_to_f16_kernel<<<(n4 + 255) / 256, 256, 0, stream>>>(W, Wh, n4);
        qmha_wmma_kernel<true><<<grid, block, 0, stream>>>(
            x, theta, W, Wh, bias, out, M, N, K);
    } else {
        qmha_wmma_kernel<false><<<grid, block, 0, stream>>>(
            x, theta, W, nullptr, bias, out, M, N, K);
    }
}
